// GNN_28389733826691
// MI455X (gfx1250) — matrix operand order repaired and checked by reading, not compiled, not run on hardware
//
#include <hip/hip_runtime.h>
#include <hip/hip_bf16.h>

// ---------------------------------------------------------------------------
// Types for CDNA5 WMMA (wave32). v_wmma_f32_16x16x32_bf16:
//   A: 16x32 bf16 (16 elts/lane), B: 32x16 bf16 (16 elts/lane), C/D: 16x16 f32
// ---------------------------------------------------------------------------
typedef __attribute__((ext_vector_type(16))) __bf16 v16bf;
typedef __attribute__((ext_vector_type(8)))  __bf16 v8bf;
typedef __attribute__((ext_vector_type(8)))  float  v8f;
typedef __attribute__((ext_vector_type(4)))  int    v4i;

__device__ inline v8f wmma_bf16(v16bf a, v16bf b, v8f c) {
    // (neg_a, A, neg_b, B, c_mod, C, reuse_a, reuse_b)
    return __builtin_amdgcn_wmma_f32_16x16x32_bf16(false, a, false, b,
                                                   (short)0, c, false, false);
}

// ---------------------------------------------------------------------------
// Async global->LDS staging (CDNA5 GLOBAL_LOAD_ASYNC_TO_LDS_B128, ASYNCcnt).
// Builtin signature (from clang diagnostic): arg0 = AS1 (global) v4i*,
// arg1 = LDS v4i*, then imm offset + imm cpol.
// Falls back to synchronous VGPR copies if the builtin is unavailable.
// ---------------------------------------------------------------------------
#if __has_builtin(__builtin_amdgcn_global_load_async_to_lds_b128)
#define HAVE_ASYNC_LDS 1
#else
#define HAVE_ASYNC_LDS 0
#endif

typedef __attribute__((address_space(1))) v4i gv4i_t;
typedef __attribute__((address_space(3))) v4i lv4i_t;

__device__ inline void async_copy_b128(const __bf16* g, __bf16* l) {
#if HAVE_ASYNC_LDS
    __builtin_amdgcn_global_load_async_to_lds_b128(
        (gv4i_t*)(g), (lv4i_t*)(l), 0, 0);
#else
    *reinterpret_cast<v8bf*>(l) = *reinterpret_cast<const v8bf*>(g);
#endif
}

__device__ inline void async_wait_all() {
#if HAVE_ASYNC_LDS
#if __has_builtin(__builtin_amdgcn_s_wait_asynccnt)
    __builtin_amdgcn_s_wait_asynccnt(0);
#else
    asm volatile("s_wait_asynccnt 0x0" ::: "memory");
#endif
#endif
}

// A fragment, 16x32 tile from row-major LDS array As (ld elements, ld % 8 == 0).
// Lane l: row = l&15; K group = (l>=16) ? 8 : 0.
// elts 0..7  -> K = kbase+0..7      (VGPR 0..3)
// elts 8..15 -> K = 16+kbase+0..7   (VGPR 4..7)
__device__ inline v16bf load_frag_a(const __bf16* As, int ld, int row0, int k0, int lane) {
    int row = row0 + (lane & 15);
    int ks  = k0 + ((lane >> 4) << 3);
    const __bf16* p = As + row * ld + ks;
    v16bf a;
    *reinterpret_cast<v8bf*>(&a)       = *reinterpret_cast<const v8bf*>(p);       // K=ks..ks+7
    *(reinterpret_cast<v8bf*>(&a) + 1) = *reinterpret_cast<const v8bf*>(p + 16);  // K=ks+16..+23
    return a;
}

// B fragment, 32x16 tile. Bt is staged transposed: Bt[n][k] (row-major, ld % 8 == 0),
// so each lane's K-values lie in two contiguous 8-element runs.
// Lane l: col n = l&15; K = ((l>=16)?8:0) + e for e=0..7, 16 + ((l>=16)?8:0) + (e-8) for e=8..15.
__device__ inline v16bf load_frag_b(const __bf16* Bt, int ld, int n0, int k0, int lane) {
    int n  = n0 + (lane & 15);
    int kb = k0 + ((lane >> 4) << 3);
    const __bf16* p = Bt + n * ld + kb;
    v16bf b;
    *reinterpret_cast<v8bf*>(&b)       = *reinterpret_cast<const v8bf*>(p);
    *(reinterpret_cast<v8bf*>(&b) + 1) = *reinterpret_cast<const v8bf*>(p + 16);
    return b;
}

// ---------------------------------------------------------------------------
// fp32 -> bf16 elementwise convert (n % 4 == 0)
// ---------------------------------------------------------------------------
__global__ __launch_bounds__(256) void k_f32_to_bf16(const float* __restrict__ src,
                                                     __bf16* __restrict__ dst, int n) {
    int i = (blockIdx.x * 256 + threadIdx.x) * 4;
    if (i + 3 < n) {
        float4 v = *reinterpret_cast<const float4*>(src + i);
        dst[i + 0] = (__bf16)v.x;
        dst[i + 1] = (__bf16)v.y;
        dst[i + 2] = (__bf16)v.z;
        dst[i + 3] = (__bf16)v.w;
    }
}

// ---------------------------------------------------------------------------
// fp32 [R,C] -> bf16 transposed [C,R]   (R,C multiples of 32)
// block (32,8), grid (C/32, R/32)
// ---------------------------------------------------------------------------
__global__ __launch_bounds__(256) void k_transpose_bf16(const float* __restrict__ src,
                                                        __bf16* __restrict__ dst,
                                                        int R, int C) {
    __shared__ __bf16 tile[32][33];
    int c0 = blockIdx.x * 32, r0 = blockIdx.y * 32;
    for (int i = threadIdx.y; i < 32; i += 8)
        tile[i][threadIdx.x] = (__bf16)src[(size_t)(r0 + i) * C + c0 + threadIdx.x];
    __syncthreads();
    for (int i = threadIdx.y; i < 32; i += 8)
        dst[(size_t)(c0 + i) * R + r0 + threadIdx.x] = tile[threadIdx.x][i];
}

// ---------------------------------------------------------------------------
// Batched: gso fp32 [B][128][128] -> Stg bf16 [B][128][128] transposed per b:
// Stg[b][m][n] = gso[b][n][m].  grid (4,4,B), block (32,8)
// ---------------------------------------------------------------------------
__global__ __launch_bounds__(256) void k_gso_to_bf16_t(const float* __restrict__ gso,
                                                       __bf16* __restrict__ Stg) {
    __shared__ __bf16 tile[32][33];
    const int b = blockIdx.z;
    const float* S = gso + (size_t)b * 128 * 128;
    __bf16* D = Stg + (size_t)b * 128 * 128;
    int c0 = blockIdx.x * 32, r0 = blockIdx.y * 32;
    for (int i = threadIdx.y; i < 32; i += 8)
        tile[i][threadIdx.x] = (__bf16)S[(r0 + i) * 128 + c0 + threadIdx.x];
    __syncthreads();
    for (int i = threadIdx.y; i < 32; i += 8)
        D[(c0 + i) * 128 + r0 + threadIdx.x] = tile[threadIdx.x][i];
}

// ---------------------------------------------------------------------------
// Generic WMMA GEMM: C[M,N] = act(A[M,K](bf16,rowmajor) * Bt[N,K]^T + bias[N])
// block = 256 threads (8 waves), 64x64 C tile, K-step 32,
// double-buffered async global->LDS staging.
// ---------------------------------------------------------------------------
template <bool RELU, typename OutT>
__global__ __launch_bounds__(256) void k_gemm(const __bf16* __restrict__ A,
                                              const __bf16* __restrict__ Bt,
                                              const float* __restrict__ bias,
                                              OutT* __restrict__ C,
                                              int M, int N, int K) {
    __shared__ __bf16 As[2][64 * 32];
    __shared__ __bf16 Bs[2][64 * 32];
    const int m0 = blockIdx.y * 64, n0 = blockIdx.x * 64;
    const int t = threadIdx.x, lane = t & 31, w = t >> 5;
    const int tr = w >> 2, tc = w & 3;                 // wave owns tiles (tr,tc) & (tr+2,tc)
    const int r  = t >> 2;                             // staging row 0..63
    const int kc = (t & 3) << 3;                       // staging K chunk (8 bf16)

    // prologue: stage K-tile 0 into buffer 0
    async_copy_b128(&A[(size_t)(m0 + r) * K + kc], &As[0][r * 32 + kc]);
    async_copy_b128(&Bt[(size_t)(n0 + r) * K + kc], &Bs[0][r * 32 + kc]);
    async_wait_all();
    __syncthreads();

    v8f acc0 = {}, acc1 = {};
    const int nk = K >> 5;
    for (int i = 0; i < nk; ++i) {
        const int cur = i & 1;
        if (i + 1 < nk) {   // prefetch next K-tile into the other buffer
            const int k0 = (i + 1) << 5;
            async_copy_b128(&A[(size_t)(m0 + r) * K + k0 + kc], &As[cur ^ 1][r * 32 + kc]);
            async_copy_b128(&Bt[(size_t)(n0 + r) * K + k0 + kc], &Bs[cur ^ 1][r * 32 + kc]);
        }
        v16bf fb  = load_frag_b(Bs[cur], 32, tc * 16, 0, lane);
        v16bf fa0 = load_frag_a(As[cur], 32, tr * 16, 0, lane);
        v16bf fa1 = load_frag_a(As[cur], 32, (tr + 2) * 16, 0, lane);
        acc0 = wmma_bf16(fa0, fb, acc0);
        acc1 = wmma_bf16(fa1, fb, acc1);
        async_wait_all();
        __syncthreads();
    }

    const int col = n0 + tc * 16 + (lane & 15);
    const float bv = bias[col];
    const int rb0 = m0 + tr * 16 + ((lane >> 4) << 3);
    const int rb1 = rb0 + 32;
    for (int i = 0; i < 8; ++i) {
        float v0 = acc0[i] + bv, v1 = acc1[i] + bv;
        if (RELU) { v0 = fmaxf(v0, 0.f); v1 = fmaxf(v1, 0.f); }
        C[(size_t)(rb0 + i) * N + col] = (OutT)v0;
        C[(size_t)(rb1 + i) * N + col] = (OutT)v1;
    }
}

// ---------------------------------------------------------------------------
// Hop kernel: per batch element b, z1 = z0 * S_b, z2 = z1 * S_b  (G in {32,64})
// S_b^T pre-converted to bf16 (Stg); staged into LDS with async b128 copies.
// ---------------------------------------------------------------------------
__global__ __launch_bounds__(256) void k_hop(const __bf16* __restrict__ z0,
                                             const __bf16* __restrict__ Stg,
                                             __bf16* __restrict__ z1,
                                             __bf16* __restrict__ z2, int G) {
    __shared__ __bf16 St[128 * 128];   // St[m][n] = S[n][m]   (32 KB)
    __shared__ __bf16 Za[64 * 128];    // current z            (16 KB)
    __shared__ __bf16 Zb[64 * 128];    // hop-1 result         (16 KB)
    const int b = blockIdx.x, t = threadIdx.x, lane = t & 31, w = t >> 5;

    const __bf16* Sg = Stg + (size_t)b * 128 * 128;
    for (int i = t * 8; i < 128 * 128; i += 256 * 8)
        async_copy_b128(&Sg[i], &St[i]);
    const __bf16* zsrc = z0 + (size_t)b * G * 128;
    for (int i = t * 8; i < G * 128; i += 256 * 8)
        async_copy_b128(&zsrc[i], &Za[i]);
    async_wait_all();
    __syncthreads();

    const int ntiles = (G >> 4) * 8;   // (G/16) row tiles x 8 col tiles
    __bf16* g1 = z1 + (size_t)b * G * 128;
    for (int idx = w; idx < ntiles; idx += 8) {
        int tr = idx >> 3, tc = idx & 7;
        v8f acc = {};
        for (int k0 = 0; k0 < 128; k0 += 32) {
            v16bf fa = load_frag_a(Za, 128, tr * 16, k0, lane);
            v16bf fb = load_frag_b(St, 128, tc * 16, k0, lane);
            acc = wmma_bf16(fa, fb, acc);
        }
        int col = tc * 16 + (lane & 15);
        int row0 = tr * 16 + ((lane >> 4) << 3);
        for (int i = 0; i < 8; ++i) {
            __bf16 v = (__bf16)acc[i];
            Zb[(row0 + i) * 128 + col] = v;
            g1[(size_t)(row0 + i) * 128 + col] = v;
        }
    }
    __syncthreads();

    __bf16* g2 = z2 + (size_t)b * G * 128;
    for (int idx = w; idx < ntiles; idx += 8) {
        int tr = idx >> 3, tc = idx & 7;
        v8f acc = {};
        for (int k0 = 0; k0 < 128; k0 += 32) {
            v16bf fa = load_frag_a(Zb, 128, tr * 16, k0, lane);
            v16bf fb = load_frag_b(St, 128, tc * 16, k0, lane);
            acc = wmma_bf16(fa, fb, acc);
        }
        int col = tc * 16 + (lane & 15);
        int row0 = tr * 16 + ((lane >> 4) << 3);
        for (int i = 0; i < 8; ++i)
            g2[(size_t)(row0 + i) * 128 + col] = (__bf16)acc[i];
    }
}

// ---------------------------------------------------------------------------
// Node-varying reduce: out[b,f,n] = act( sum_{k,g} z_k[b,g,n]*h[f,k,g,n] + bias[f] )
// Per block: one graph node n, 64 batch rows. GEMM [64,3G] x [3G,F] via WMMA.
// grid = (128, B/64). KG in {96,192}; F in {32,64}.
// ---------------------------------------------------------------------------
__global__ __launch_bounds__(256) void k_reduce(const __bf16* __restrict__ z0,
                                                const __bf16* __restrict__ z1,
                                                const __bf16* __restrict__ z2,
                                                const float* __restrict__ h,
                                                const float* __restrict__ bias,
                                                __bf16* __restrict__ out,
                                                int G, int F, int relu) {
    __shared__ __bf16 Hs[64 * 192];    // Bt[f][kg]  (<=24 KB)
    __shared__ __bf16 As[64 * 192];    // A[row][kg] (<=24 KB)
    const int n = blockIdx.x, b0 = blockIdx.y * 64;
    const int KG = 3 * G;
    const int t = threadIdx.x, lane = t & 31, w = t >> 5;

    for (int i = t; i < F * KG; i += 256) {
        int f = i / KG, kg = i - f * KG;
        int k = kg / G, g = kg - k * G;
        Hs[f * KG + kg] = (__bf16)h[(((size_t)f * 3 + k) * G + g) * 128 + n];
    }
    for (int i = t; i < 64 * KG; i += 256) {
        int r = i / KG, kg = i - r * KG;
        int k = kg / G, g = kg - k * G;
        const __bf16* zp = (k == 0) ? z0 : (k == 1) ? z1 : z2;
        As[r * KG + kg] = zp[((size_t)(b0 + r) * G + g) * 128 + n];
    }
    __syncthreads();

    const int ct = F >> 4;             // col tiles (2 or 4)
    const int ntiles = 4 * ct;         // 4 row tiles
    for (int idx = w; idx < ntiles; idx += 8) {
        int tr = idx / ct, tc = idx - tr * ct;
        v8f acc = {};
        for (int k0 = 0; k0 < KG; k0 += 32) {
            v16bf fa = load_frag_a(As, KG, tr * 16, k0, lane);
            v16bf fb = load_frag_b(Hs, KG, tc * 16, k0, lane);
            acc = wmma_bf16(fa, fb, acc);
        }
        int fcol = tc * 16 + (lane & 15);
        float bv = bias[fcol];
        int row0 = tr * 16 + ((lane >> 4) << 3);
        for (int i = 0; i < 8; ++i) {
            float v = acc[i] + bv;
            if (relu) v = fmaxf(v, 0.f);
            out[((size_t)(b0 + row0 + i) * F + fcol) * 128 + n] = (__bf16)v;
        }
    }
}

// ---------------------------------------------------------------------------
// Host-side orchestration
// ---------------------------------------------------------------------------
extern "C" void kernel_launch(void* const* d_in, const int* in_sizes, int n_in,
                              void* d_out, int out_size, void* d_ws, size_t ws_size,
                              hipStream_t stream) {
    (void)in_sizes; (void)n_in; (void)out_size; (void)ws_size;
    constexpr int B = 4096, N = 128;

    const float* x     = (const float*)d_in[0];   // [B,8,128]  -> [B,1024]
    const float* gso   = (const float*)d_in[1];   // [B,128,128]
    const float* W_enc = (const float*)d_in[2];   // [1024,4096]
    const float* b_enc = (const float*)d_in[3];   // [4096]
    const float* h1    = (const float*)d_in[4];   // [64,3,32,128]
    const float* bias1 = (const float*)d_in[5];   // [64]
    const float* h2    = (const float*)d_in[6];   // [64,3,64,128]
    const float* bias2 = (const float*)d_in[7];   // [64]
    const float* h3    = (const float*)d_in[8];   // [32,3,64,128]
    const float* bias3 = (const float*)d_in[9];   // [32]
    const float* W_dec = (const float*)d_in[10];  // [4096,2048]
    const float* b_dec = (const float*)d_in[11];  // [2048]
    float* outp = (float*)d_out;                  // [B,16,128]

    char* ws = (char*)d_ws;
    constexpr size_t SZ_XE = (size_t)4096 * 1024 * 2;       // 8 MiB
    constexpr size_t SZ_WE = (size_t)4096 * 1024 * 2;       // 8 MiB (WencT [4096,1024])
    constexpr size_t SZ_WD = (size_t)2048 * 4096 * 2;       // 16 MiB (WdecT [2048,4096])
    constexpr size_t SZ_Z  = (size_t)4096 * 64 * 128 * 2;   // 64 MiB
    __bf16* Xe    = (__bf16*)(ws);
    __bf16* WencT = (__bf16*)(ws + SZ_XE);
    __bf16* WdecT = (__bf16*)(ws + SZ_XE + SZ_WE);
    __bf16* zA    = (__bf16*)(ws + SZ_XE + SZ_WE + SZ_WD);
    __bf16* z1    = (__bf16*)(ws + SZ_XE + SZ_WE + SZ_WD + SZ_Z);
    __bf16* z2    = (__bf16*)(ws + SZ_XE + SZ_WE + SZ_WD + 2 * SZ_Z);
    __bf16* zB    = (__bf16*)(ws + SZ_XE + SZ_WE + SZ_WD + 3 * SZ_Z);
    __bf16* Stg   = (__bf16*)(ws + SZ_XE + SZ_WE + SZ_WD + 4 * SZ_Z);  // 128 MiB

    // 1) convert inputs (x, W_enc^T, W_dec^T, gso^T as bf16)
    k_f32_to_bf16<<<(B * 1024) / (256 * 4), 256, 0, stream>>>(x, Xe, B * 1024);
    k_transpose_bf16<<<dim3(4096 / 32, 1024 / 32), dim3(32, 8), 0, stream>>>(W_enc, WencT, 1024, 4096);
    k_transpose_bf16<<<dim3(2048 / 32, 4096 / 32), dim3(32, 8), 0, stream>>>(W_dec, WdecT, 4096, 2048);
    k_gso_to_bf16_t<<<dim3(4, 4, B), dim3(32, 8), 0, stream>>>(gso, Stg);

    // 2) encoder: [B,1024] x [1024,4096] + b_enc, ReLU -> zA = xg1 [B,32,128] bf16
    k_gemm<true, __bf16><<<dim3(4096 / 64, B / 64), 256, 0, stream>>>(Xe, WencT, b_enc, zA, B, 4096, 1024);

    // 3) layer 1 (G=32 -> F=64, ReLU)
    k_hop<<<B, 256, 0, stream>>>(zA, Stg, z1, z2, 32);
    k_reduce<<<dim3(N, B / 64), 256, 0, stream>>>(zA, z1, z2, h1, bias1, zB, 32, 64, 1);

    // 4) layer 2 (G=64 -> F=64, ReLU)
    k_hop<<<B, 256, 0, stream>>>(zB, Stg, z1, z2, 64);
    k_reduce<<<dim3(N, B / 64), 256, 0, stream>>>(zB, z1, z2, h2, bias2, zA, 64, 64, 1);

    // 5) layer 3 (G=64 -> F=32, no activation)
    k_hop<<<B, 256, 0, stream>>>(zA, Stg, z1, z2, 64);
    k_reduce<<<dim3(N, B / 64), 256, 0, stream>>>(zA, z1, z2, h3, bias3, zB, 64, 32, 0);

    // 6) decoder: [B,4096] x [4096,2048] + b_dec -> d_out fp32 [B,16,128]
    k_gemm<false, float><<<dim3(2048 / 64, B / 64), 256, 0, stream>>>(zB, WdecT, b_dec, outp, B, 2048, 4096);
}
